// Conditional_Contrastive_loss_77988016161230
// MI455X (gfx1250) — matrix operand order repaired and checked by reading, not compiled, not run on hardware
//
#include <hip/hip_runtime.h>

#define NROWS 8192
#define DIM   128
#define NCLS  1000
#define EPSN  1e-8f
#define BT    128          // block tile (M and N)
#define KP    136          // padded halves per LDS row: 272B rows, 16B aligned

typedef __attribute__((ext_vector_type(16))) _Float16 v16h;
typedef __attribute__((ext_vector_type(4)))  _Float16 v4h;
typedef __attribute__((ext_vector_type(8)))  float    v8f;

union FragAB { v16h v; uint4 q[2]; unsigned u[8]; };
union FragC  { v8f  v; float f[8]; };

__device__ inline float wave_allreduce_add(float x) {
#pragma unroll
  for (int off = 16; off >= 1; off >>= 1) x += __shfl_xor(x, off, 32);
  return x;
}

// ---------------------------------------------------------------------------
// init: zero S_all, S_pos, out
// ---------------------------------------------------------------------------
__global__ void init_kernel(float* __restrict__ Sall, float* __restrict__ Spos,
                            float* __restrict__ out) {
  int i = blockIdx.x * blockDim.x + threadIdx.x;
  if (i < NROWS) { Sall[i] = 0.0f; Spos[i] = 0.0f; }
  if (i == 0) out[0] = 0.0f;
}

// ---------------------------------------------------------------------------
// prep: one wave per row. L2-normalize, split into f16 hi/lo, proxy-exp term.
// ---------------------------------------------------------------------------
__global__ void prep_kernel(const float* __restrict__ X, const float* __restrict__ P,
                            const float* __restrict__ tptr, const float* __restrict__ mptr,
                            _Float16* __restrict__ Xh, _Float16* __restrict__ Xl,
                            float* __restrict__ pexp) {
  const int wave = threadIdx.x >> 5;
  const int lane = threadIdx.x & 31;
  const int row  = blockIdx.x * 8 + wave;
  if (row >= NROWS) return;

  const float4 x4 = *(const float4*)(X + (size_t)row * DIM + lane * 4);
  float ss = x4.x * x4.x + x4.y * x4.y + x4.z * x4.z + x4.w * x4.w;
  ss = wave_allreduce_add(ss);
  const float invn = 1.0f / fmaxf(sqrtf(ss), EPSN);
  float xn[4] = { x4.x * invn, x4.y * invn, x4.z * invn, x4.w * invn };

  const float4 p4 = *(const float4*)(P + (size_t)row * DIM + lane * 4);
  float ps = p4.x * p4.x + p4.y * p4.y + p4.z * p4.z + p4.w * p4.w;
  ps = wave_allreduce_add(ps);
  const float invp = 1.0f / fmaxf(sqrtf(ps), EPSN);
  float dot = xn[0] * p4.x * invp + xn[1] * p4.y * invp +
              xn[2] * p4.z * invp + xn[3] * p4.w * invp;
  dot = wave_allreduce_add(dot);

  v4h hv, lv;
#pragma unroll
  for (int k = 0; k < 4; ++k) {
    _Float16 h = (_Float16)xn[k];
    hv[k] = h;
    lv[k] = (_Float16)(xn[k] - (float)h);
  }
  *(v4h*)(Xh + (size_t)row * DIM + lane * 4) = hv;
  *(v4h*)(Xl + (size_t)row * DIM + lane * 4) = lv;

  if (lane == 0) {
    const float invT = 1.0f / tptr[0];
    pexp[row] = __expf((dot - mptr[0]) * invT);
  }
}

// ---------------------------------------------------------------------------
// mask transpose: nm[N][C] f32 -> MT[C][N] u8 (0/1), LDS-tiled
// ---------------------------------------------------------------------------
__global__ void mask_transpose_kernel(const float* __restrict__ nm,
                                      unsigned char* __restrict__ MT) {
  __shared__ float tile[32][33];
  const int cb = blockIdx.x * 32, jb = blockIdx.y * 32;
  const int tx = threadIdx.x, ty = threadIdx.y;
  const int c = cb + tx, j = jb + ty;
  tile[ty][tx] = (c < NCLS) ? nm[(size_t)j * NCLS + c] : 0.0f;
  __syncthreads();
  const int co = cb + ty, jo = jb + tx;
  if (co < NCLS)
    MT[(size_t)co * NROWS + jo] = (tile[tx][ty] != 0.0f) ? (unsigned char)1 : (unsigned char)0;
}

// ---------------------------------------------------------------------------
// main: triangular-blocked split-f16 WMMA Gram matrix + fused exp/mask epilogue
// ---------------------------------------------------------------------------
__global__ __launch_bounds__(256) void sim_loss_kernel(
    const _Float16* __restrict__ Xh, const _Float16* __restrict__ Xl,
    const int* __restrict__ labels, const unsigned char* __restrict__ MT,
    const float* __restrict__ tptr, const float* __restrict__ mptr,
    float* __restrict__ Sall, float* __restrict__ Spos) {
  __shared__ _Float16 Ah[BT][KP];
  __shared__ _Float16 Al[BT][KP];
  __shared__ _Float16 Bh[BT][KP];
  __shared__ _Float16 Bl[BT][KP];
  __shared__ int lblA[BT];
  __shared__ int lblB[BT];

  // decode upper-triangular block (br <= bc)
  const int NTI = NROWS / BT;   // 64
  int id = blockIdx.x;
  int br = 0;
  while (id >= NTI - br) { id -= NTI - br; ++br; }
  const int bc = br + id;

  const int tid   = threadIdx.x;
  const int rowA0 = br * BT, rowB0 = bc * BT;

  // cooperative b128 global -> b128 LDS staging (128 rows x 16 uint4 per array)
  {
    const uint4* gAh = (const uint4*)(Xh + (size_t)rowA0 * DIM);
    const uint4* gAl = (const uint4*)(Xl + (size_t)rowA0 * DIM);
    const uint4* gBh = (const uint4*)(Xh + (size_t)rowB0 * DIM);
    const uint4* gBl = (const uint4*)(Xl + (size_t)rowB0 * DIM);
#pragma unroll
    for (int it = 0; it < 8; ++it) {
      int idx = it * 256 + tid;      // 0..2047
      int r   = idx >> 4;            // row 0..127
      int cq  = idx & 15;            // uint4 within row
      *(uint4*)&Ah[r][cq * 8] = gAh[(size_t)r * 16 + cq];
      *(uint4*)&Al[r][cq * 8] = gAl[(size_t)r * 16 + cq];
      *(uint4*)&Bh[r][cq * 8] = gBh[(size_t)r * 16 + cq];
      *(uint4*)&Bl[r][cq * 8] = gBl[(size_t)r * 16 + cq];
    }
    if (tid < BT)            lblA[tid]      = labels[rowA0 + tid];
    else if (tid < 2 * BT)   lblB[tid - BT] = labels[rowB0 + tid - BT];
  }
  __syncthreads();

  const int lane  = tid & 31;
  const int wv    = tid >> 5;      // 0..7
  const int waveM = wv >> 1;       // 0..3 : 32-row slab
  const int waveN = wv & 1;        // 0..1 : 64-col slab
  const int l16   = lane & 15;
  const int hlf   = lane >> 4;

  FragC acc[2][4];
#pragma unroll
  for (int mt = 0; mt < 2; ++mt)
#pragma unroll
    for (int nt = 0; nt < 4; ++nt)
#pragma unroll
      for (int r = 0; r < 8; ++r) acc[mt][nt].f[r] = 0.0f;

#pragma unroll
  for (int kc = 0; kc < DIM / 32; ++kc) {
    const int kbase = kc * 32;
    FragAB aH[2], aL[2], bH[4], bL[4];
    // A fragment (16x32 f16): one contiguous 32B run per lane -> 2x ds_load_b128
#pragma unroll
    for (int mt = 0; mt < 2; ++mt) {
      const int rr = waveM * 32 + mt * 16 + l16;
      const _Float16* pAh = &Ah[rr][kbase + hlf * 16];
      const _Float16* pAl = &Al[rr][kbase + hlf * 16];
      aH[mt].q[0] = *(const uint4*)pAh;
      aH[mt].q[1] = *(const uint4*)(pAh + 8);
      aL[mt].q[0] = *(const uint4*)pAl;
      aL[mt].q[1] = *(const uint4*)(pAl + 8);
    }
    // B fragment (32x16 f16): one contiguous 32B run per lane -> 2x ds_load_b128
#pragma unroll
    for (int nt = 0; nt < 4; ++nt) {
      const int rr = waveN * 64 + nt * 16 + l16;
      const _Float16* pBh = &Bh[rr][kbase + hlf * 16];
      const _Float16* pBl = &Bl[rr][kbase + hlf * 16];
      bH[nt].q[0] = *(const uint4*)pBh;
      bH[nt].q[1] = *(const uint4*)(pBh + 8);
      bL[nt].q[0] = *(const uint4*)pBl;
      bL[nt].q[1] = *(const uint4*)(pBl + 8);
    }
    // split-f16 product: hi*hi + hi*lo + lo*hi  (near-fp32 cosine similarity)
#pragma unroll
    for (int mt = 0; mt < 2; ++mt)
#pragma unroll
      for (int nt = 0; nt < 4; ++nt) {
        acc[mt][nt].v = __builtin_amdgcn_wmma_f32_16x16x32_f16(
            false, aH[mt].v, false, bH[nt].v, (short)0, acc[mt][nt].v, false, false);
        acc[mt][nt].v = __builtin_amdgcn_wmma_f32_16x16x32_f16(
            false, aH[mt].v, false, bL[nt].v, (short)0, acc[mt][nt].v, false, false);
        acc[mt][nt].v = __builtin_amdgcn_wmma_f32_16x16x32_f16(
            false, aL[mt].v, false, bH[nt].v, (short)0, acc[mt][nt].v, false, false);
      }
  }

  const float invT   = 1.0f / tptr[0];
  const float mrg    = mptr[0];
  const bool  mirror = (br != bc);

  float cs[4]  = {0.0f, 0.0f, 0.0f, 0.0f};   // column partials (across mt, r)
  float mcs[4] = {0.0f, 0.0f, 0.0f, 0.0f};

#pragma unroll
  for (int mt = 0; mt < 2; ++mt) {
    const int rloc  = waveM * 32 + mt * 16 + hlf * 8;
    const int gRow0 = rowA0 + rloc;
    int cA[8];
#pragma unroll
    for (int r = 0; r < 8; ++r) cA[r] = lblA[rloc + r];

    float rs[8] = {0,0,0,0,0,0,0,0};
    float ms[8] = {0,0,0,0,0,0,0,0};

#pragma unroll
    for (int nt = 0; nt < 4; ++nt) {
      const int cloc = waveN * 64 + nt * 16 + l16;
      const int gCol = rowB0 + cloc;

      float e[8];
#pragma unroll
      for (int r = 0; r < 8; ++r) {
        float ev = __expf((acc[mt][nt].f[r] - mrg) * invT);
        if (gRow0 + r == gCol) ev = 0.0f;                 // remove diagonal
        e[r] = ev;
      }

      // row-direction: label uniform per row -> 16B coalesced mask reads
#pragma unroll
      for (int r = 0; r < 8; ++r) {
        const float mk = (float)MT[(size_t)cA[r] * NROWS + gCol];
        rs[r] += e[r];
        ms[r] += e[r] * mk;
      }

      // mirrored (symmetry): 8 consecutive mask bytes -> single 8B load
      if (mirror) {
        const int cB = lblB[cloc];
        const uint2 mb = *(const uint2*)(MT + (size_t)cB * NROWS + gRow0);
#pragma unroll
        for (int r = 0; r < 8; ++r) {
          const unsigned w = (r < 4) ? mb.x : mb.y;
          const float mk = (float)((w >> (8 * (r & 3))) & 0xffu);
          cs[nt]  += e[r];
          mcs[nt] += e[r] * mk;
        }
      }
    }

    // reduce row sums across the 16 lanes of each half, then atomics
#pragma unroll
    for (int off = 8; off >= 1; off >>= 1)
#pragma unroll
      for (int r = 0; r < 8; ++r) {
        rs[r] += __shfl_xor(rs[r], off, 32);
        ms[r] += __shfl_xor(ms[r], off, 32);
      }
    if (l16 == 0) {
#pragma unroll
      for (int r = 0; r < 8; ++r) {
        atomicAdd(&Sall[gRow0 + r], rs[r]);
        atomicAdd(&Spos[gRow0 + r], ms[r]);
      }
    }
  }

  if (mirror) {
#pragma unroll
    for (int nt = 0; nt < 4; ++nt) {
      const float c = cs[nt]  + __shfl_xor(cs[nt], 16, 32);
      const float m = mcs[nt] + __shfl_xor(mcs[nt], 16, 32);
      if (hlf == 0) {
        const int gCol = rowB0 + waveN * 64 + nt * 16 + l16;
        atomicAdd(&Sall[gCol], c);
        atomicAdd(&Spos[gCol], m);
      }
    }
  }
}

// ---------------------------------------------------------------------------
// finalize: loss_i = -log(T*(p+Spos)/(p+Sall)), mean over N
// ---------------------------------------------------------------------------
__global__ void finalize_kernel(const float* __restrict__ pexp,
                                const float* __restrict__ Sall,
                                const float* __restrict__ Spos,
                                const float* __restrict__ tptr,
                                float* __restrict__ out) {
  const float T = tptr[0];
  float local = 0.0f;
  for (int i = blockIdx.x * blockDim.x + threadIdx.x; i < NROWS;
       i += gridDim.x * blockDim.x) {
    const float p = pexp[i];
    local += -logf(T * (p + Spos[i]) / (p + Sall[i]));
  }
  local = wave_allreduce_add(local);
  if ((threadIdx.x & 31) == 0) atomicAdd(out, local * (1.0f / (float)NROWS));
}

// ---------------------------------------------------------------------------
extern "C" void kernel_launch(void* const* d_in, const int* in_sizes, int n_in,
                              void* d_out, int out_size, void* d_ws, size_t ws_size,
                              hipStream_t stream) {
  const float* X  = (const float*)d_in[0];   // inst_embed [N,D]
  const float* P  = (const float*)d_in[1];   // proxy      [N,D]
  const float* NM = (const float*)d_in[2];   // negative_mask [N,C] f32
  const int*   LB = (const int*)d_in[3];     // labels [N]
  const float* Tp = (const float*)d_in[4];   // temperature
  const float* Mg = (const float*)d_in[5];   // margin
  float* out = (float*)d_out;

  char* ws = (char*)d_ws;
  _Float16* Xh = (_Float16*)ws;                                   // 2 MB
  _Float16* Xl = (_Float16*)(ws + (size_t)NROWS * DIM * 2);       // 2 MB
  float* pexp  = (float*)(ws + (size_t)NROWS * DIM * 4);
  float* Sall  = pexp + NROWS;
  float* Spos  = Sall + NROWS;
  unsigned char* MT = (unsigned char*)(Spos + NROWS);             // 8 MB

  init_kernel<<<(NROWS + 255) / 256, 256, 0, stream>>>(Sall, Spos, out);
  prep_kernel<<<NROWS / 8, 256, 0, stream>>>(X, P, Tp, Mg, Xh, Xl, pexp);
  dim3 tb(32, 32);
  dim3 tg((NCLS + 31) / 32, NROWS / 32);
  mask_transpose_kernel<<<tg, tb, 0, stream>>>(NM, MT);

  const int nTiles  = NROWS / BT;                    // 64
  const int nBlocks = nTiles * (nTiles + 1) / 2;     // 2080
  sim_loss_kernel<<<nBlocks, 256, 0, stream>>>(Xh, Xl, LB, MT, Tp, Mg, Sall, Spos);

  finalize_kernel<<<32, 256, 0, stream>>>(pexp, Sall, Spos, Tp, out);
}
